// TemporalAttention_A2V_84241488544293
// MI455X (gfx1250) — matrix operand order repaired and checked by reading, not compiled, not run on hardware
//
#include <hip/hip_runtime.h>
#include <hip/hip_bf16.h>

// ---------------------------------------------------------------------------
// TemporalAttention A2V (Mamba-gated multi-scale feature modulation), gfx1250.
//
// Roofline: elementwise gating moves ~683MB HBM traffic -> ~29us floor at
// 23.3 TB/s. The Mamba stack is <1 GFLOP on an 80x256 token tensor, fully
// L2-resident; it is executed by a single persistent kernel (4 blocks, one
// per branch) whose dense GEMMs use V_WMMA_F32_16X16X32_F16.
// ---------------------------------------------------------------------------

#define NW  32              // waves per block in the mamba kernel (wave32)
#define BLK (NW * 32)       // 1024 threads

typedef _Float16 v16h __attribute__((ext_vector_type(16)));
typedef float    v8f  __attribute__((ext_vector_type(8)));

__device__ __forceinline__ float siluf(float x) { return x / (1.0f + __expf(-x)); }

// ---- WMMA fragment loaders (CDNA5 ISA 7.12.2 layouts, wave32) --------------
// A: 16x32 f16, one M-row per lane (lanes 0-15 & 16-31 hold same rows,
//    different K halves): v0..3 -> K = 2v+w + 8*(lane>=16); v4..7 -> K = 16+...
__device__ __forceinline__ v16h load_a_frag(const float* __restrict__ A, int lda,
                                            int r0, int k0, int kvalid, int lane) {
  const float* row = A + (size_t)(r0 + (lane & 15)) * lda + k0;
  const int kh = ((lane >> 4) & 1) << 3;   // 0 or 8
  v16h a;
#pragma unroll
  for (int j = 0; j < 16; ++j) {
    const int v = j >> 1, w = j & 1;
    const int k = (v < 4) ? (v * 2 + w + kh) : (16 + (v - 4) * 2 + w + kh);
    a[j] = (k < kvalid) ? (_Float16)row[k] : (_Float16)0.0f;
  }
  return a;
}

// B: 32x16 f16 (KxN), one N-col per lane; same K order as A:
//    j<8 -> K = j + 8*(lane>=16); j>=8 -> K = 16+(j-8)+8*(lane>=16). Weights stored row-major as W[N][K].
__device__ __forceinline__ v16h load_b_frag(const float* __restrict__ W, int ldw,
                                            int n0, int k0, int kvalid, int lane) {
  const float* row = W + (size_t)(n0 + (lane & 15)) * ldw + k0;
  const int kh = ((lane >> 4) & 1) << 3;   // 0 or 8
  v16h b;
#pragma unroll
  for (int j = 0; j < 16; ++j) {
    const int k = (j < 8) ? (kh + j) : (16 + kh + (j - 8));
    b[j] = (k < kvalid) ? (_Float16)row[k] : (_Float16)0.0f;
  }
  return b;
}

// C/D: v8f, col = n0 + lane%16, row = r0 + r + 8*(lane>=16)
__device__ __forceinline__ void store_c_frag(float* __restrict__ C, int ldc,
                                             int r0, int n0, int lane, v8f c) {
  float* col = C + (n0 + (lane & 15));
  const int rb = r0 + (((lane >> 4) & 1) << 3);
#pragma unroll
  for (int r = 0; r < 8; ++r)
    col[(size_t)(rb + r) * ldc] = c[r];
}

// C(MxN) = A(MxK) @ W(NxK)^T ; M,N multiples of 16; K zero-padded to 32.
// Tile loop is wave-uniform -> EXEC stays all-ones around v_wmma (ISA req).
__device__ void gemm_wmma(const float* __restrict__ A, int lda,
                          const float* __restrict__ W, int ldw,
                          float* __restrict__ C, int ldc,
                          int M, int N, int K, int wave, int lane) {
  const int nt = N >> 4;
  const int tiles = (M >> 4) * nt;
  const int ksteps = (K + 31) >> 5;
  for (int t = wave; t < tiles; t += NW) {
    const int r0 = (t / nt) << 4;
    const int n0 = (t % nt) << 4;
    v8f acc = {};
    for (int ks = 0; ks < ksteps; ++ks) {
      const int k0 = ks << 5;
      v16h af = load_a_frag(A, lda, r0, k0, K - k0, lane);
      v16h bf = load_b_frag(W, ldw, n0, k0, K - k0, lane);
      acc = __builtin_amdgcn_wmma_f32_16x16x32_f16(false, af, false, bf,
                                                   (short)0, acc, false, false);
    }
    store_c_frag(C, ldc, r0, n0, lane, acc);
  }
}

// ---------------------------------------------------------------------------
// Persistent per-branch Mamba pipeline: 3 layers (LN -> in_proj -> conv+silu
// -> x_proj -> dt -> scan -> gate(z) -> out_proj -> residual), then gate GEMM.
// Workspace per branch: 229120 floats; gates after that: 4 x 20480 floats.
// ---------------------------------------------------------------------------
__global__ void __launch_bounds__(BLK)
mamba_pipeline_kernel(const float* __restrict__ audio,
                      const float* __restrict__ ln_g, const float* __restrict__ ln_b,
                      const float* __restrict__ Win,  const float* __restrict__ Wc,
                      const float* __restrict__ bc,   const float* __restrict__ Wx,
                      const float* __restrict__ Wdt,  const float* __restrict__ bdt,
                      const float* __restrict__ Alog, const float* __restrict__ Dp,
                      const float* __restrict__ Wout, const float* __restrict__ Wg,
                      const float* __restrict__ bg,   float* __restrict__ ws) {
  const int br   = blockIdx.x;
  const int tid  = threadIdx.x;
  const int wave = tid >> 5, lane = tid & 31;

  float* a    = ws + (size_t)br * 229120;   // 80 x 256
  float* xz   = a   + 20480;                // 80 x 1024 (later reused 80x256)
  float* xc   = xz  + 81920;                // 80 x 512
  float* dbl  = xc  + 40960;                // 80 x 48
  float* dtb  = dbl + 3840;                 // 80 x 512
  float* yb   = dtb + 40960;                // 80 x 512
  float* gate = ws + 4 * 229120 + (size_t)br * 20480;

  for (int i = tid; i < 20480; i += BLK) a[i] = audio[i];
  __syncthreads();

  for (int l = 0; l < 3; ++l) {
    const int pl = br * 3 + l;
    const float* lg   = ln_g + pl * 256;
    const float* lb   = ln_b + pl * 256;
    const float* W1   = Win  + (size_t)pl * 1024 * 256;
    const float* wc   = Wc   + (size_t)pl * 512 * 4;
    const float* bcv  = bc   + (size_t)pl * 512;
    const float* wx   = Wx   + (size_t)pl * 48 * 512;
    const float* wdt  = Wdt  + (size_t)pl * 512 * 16;
    const float* bdtv = bdt  + (size_t)pl * 512;
    const float* alg  = Alog + (size_t)pl * 512 * 16;
    const float* dpv  = Dp   + (size_t)pl * 512;
    const float* wo   = Wout + (size_t)pl * 256 * 512;

    // -- LayerNorm (in place; one wave per token, wave32 shuffle reduction) --
    for (int t = wave; t < 80; t += NW) {
      float* row = a + t * 256;
      float s = 0.f;
      for (int j = lane; j < 256; j += 32) s += row[j];
#pragma unroll
      for (int o = 16; o; o >>= 1) s += __shfl_xor(s, o, 32);
      const float mu = s * (1.f / 256.f);
      float v = 0.f;
      for (int j = lane; j < 256; j += 32) { float d = row[j] - mu; v += d * d; }
#pragma unroll
      for (int o = 16; o; o >>= 1) v += __shfl_xor(v, o, 32);
      const float inv = rsqrtf(v * (1.f / 256.f) + 1e-5f);
      for (int j = lane; j < 256; j += 32)
        row[j] = (row[j] - mu) * inv * lg[j] + lb[j];
    }
    __syncthreads();

    // -- in_proj: xz(80x1024) = a(80x256) @ Win^T --------------------------
    gemm_wmma(a, 256, W1, 256, xz, 1024, 80, 1024, 256, wave, lane);
    __syncthreads();

    // -- depthwise causal conv (k=4) + SiLU --------------------------------
    for (int i = tid; i < 80 * 512; i += BLK) {
      const int p = i >> 9, c = i & 511;
      const int b = p / 5, t = p - b * 5;
      float acc = bcv[c];
#pragma unroll
      for (int k = 0; k < 4; ++k) {
        const int tt = t - 3 + k;
        const float xv = (tt >= 0) ? xz[(size_t)(b * 5 + tt) * 1024 + c] : 0.f;
        acc += wc[c * 4 + k] * xv;
      }
      xc[i] = siluf(acc);
    }
    __syncthreads();

    // -- x_proj: dbl(80x48) = xc @ Wx^T ------------------------------------
    gemm_wmma(xc, 512, wx, 512, dbl, 48, 80, 48, 512, wave, lane);
    __syncthreads();

    // -- dt: dtb(80x512) = softplus(dbl[:,:16] @ Wdt^T + bdt) (K=16 padded) -
    gemm_wmma(dbl, 48, wdt, 16, dtb, 512, 80, 512, 16, wave, lane);
    __syncthreads();
    for (int i = tid; i < 80 * 512; i += BLK) {
      const float x = dtb[i] + bdtv[i & 511];
      dtb[i] = (x > 20.f) ? x : log1pf(__expf(x));
    }
    __syncthreads();

    // -- selective scan over T=5 (state 16) --------------------------------
    for (int i = tid; i < 16 * 512; i += BLK) {
      const int b = i >> 9, d = i & 511;
      float An[16], h[16];
#pragma unroll
      for (int n = 0; n < 16; ++n) { An[n] = -__expf(alg[d * 16 + n]); h[n] = 0.f; }
      for (int t = 0; t < 5; ++t) {
        const int p = b * 5 + t;
        const float dtv = dtb[(size_t)p * 512 + d];
        const float xv  = xc[(size_t)p * 512 + d];
        const float* Bm = dbl + (size_t)p * 48 + 16;
        const float* Cm = dbl + (size_t)p * 48 + 32;
        float yv = 0.f;
#pragma unroll
        for (int n = 0; n < 16; ++n) {
          h[n] = __expf(dtv * An[n]) * h[n] + dtv * Bm[n] * xv;
          yv += h[n] * Cm[n];
        }
        yb[(size_t)p * 512 + d] = yv;
      }
    }
    __syncthreads();

    // -- y = (y + xc*D) * silu(z) ------------------------------------------
    for (int i = tid; i < 80 * 512; i += BLK) {
      const int p = i >> 9, c = i & 511;
      const float zv = xz[(size_t)p * 1024 + 512 + c];
      yb[i] = (yb[i] + xc[i] * dpv[c]) * siluf(zv);
    }
    __syncthreads();

    // -- out_proj into xz (reused as 80x256 scratch) + residual ------------
    gemm_wmma(yb, 512, wo, 512, xz, 256, 80, 256, 512, wave, lane);
    __syncthreads();
    for (int i = tid; i < 20480; i += BLK) a[i] += xz[i];
    __syncthreads();
  }

  // -- gate = silu(a @ gate_w^T + gate_b) ----------------------------------
  gemm_wmma(a, 256, Wg + (size_t)br * 256 * 256, 256, gate, 256, 80, 256, 256,
            wave, lane);
  __syncthreads();
  for (int i = tid; i < 20480; i += BLK)
    gate[i] = siluf(gate[i] + bg[br * 256 + (i & 255)]);
}

// ---------------------------------------------------------------------------
// Bandwidth-bound gating: o = x * (1 + g[p,c]), gate broadcast over HxW.
// blockIdx.y = plane (p*256+c) -> no integer division, uniform gate scalar.
// ---------------------------------------------------------------------------
__global__ void gate_apply_vec4(const float4* __restrict__ x, float4* __restrict__ o,
                                const float* __restrict__ g, int hw4, int transposed) {
  const int plane = blockIdx.y;                       // p*256 + c, p in [0,80)
  const int j = blockIdx.x * blockDim.x + threadIdx.x;
  if (j >= hw4) return;
  const int p = plane >> 8, c = plane & 255;
  const int row = transposed ? ((p & 15) * 5 + (p >> 4)) : p;
  const float s = 1.0f + g[row * 256 + c];
  const size_t idx = (size_t)plane * hw4 + j;
  float4 v = x[idx];
  v.x *= s; v.y *= s; v.z *= s; v.w *= s;
  o[idx] = v;
}

__global__ void gate_apply_scalar(const float* __restrict__ x, float* __restrict__ o,
                                  const float* __restrict__ g, int hw, int transposed) {
  const int plane = blockIdx.y;
  const int j = blockIdx.x * blockDim.x + threadIdx.x;
  if (j >= hw) return;
  const int p = plane >> 8, c = plane & 255;
  const int row = transposed ? ((p & 15) * 5 + (p >> 4)) : p;
  const float s = 1.0f + g[row * 256 + c];
  const size_t idx = (size_t)plane * hw + j;
  o[idx] = x[idx] * s;
}

__global__ void copy_kernel(const float* __restrict__ src, float* __restrict__ dst,
                            int n) {
  const int i = blockIdx.x * blockDim.x + threadIdx.x;
  if (i < n) dst[i] = src[i];
}

// ---------------------------------------------------------------------------
extern "C" void kernel_launch(void* const* d_in, const int* in_sizes, int n_in,
                              void* d_out, int out_size, void* d_ws, size_t ws_size,
                              hipStream_t stream) {
  const float* x1    = (const float*)d_in[0];   // (80,256,56,56)
  const float* x2    = (const float*)d_in[1];   // (80,256,28,28)
  const float* x3    = (const float*)d_in[2];   // (80,256,14,14)
  const float* x4    = (const float*)d_in[3];   // (80,256,7,7)
  const float* audio = (const float*)d_in[4];   // (16,5,256)
  const float* ln_g  = (const float*)d_in[5];
  const float* ln_b  = (const float*)d_in[6];
  const float* Win   = (const float*)d_in[7];
  const float* Wc    = (const float*)d_in[8];
  const float* bc    = (const float*)d_in[9];
  const float* Wx    = (const float*)d_in[10];
  const float* Wdt   = (const float*)d_in[11];
  const float* bdt   = (const float*)d_in[12];
  const float* Alog  = (const float*)d_in[13];
  const float* Dp    = (const float*)d_in[14];
  const float* Wout  = (const float*)d_in[15];
  const float* Wg    = (const float*)d_in[16];
  const float* bg    = (const float*)d_in[17];
  float* out = (float*)d_out;
  float* ws  = (float*)d_ws;

  // Stage 1: full Mamba stack + gate GEMMs (4 branches in parallel, WMMA).
  mamba_pipeline_kernel<<<4, BLK, 0, stream>>>(audio, ln_g, ln_b, Win, Wc, bc,
                                               Wx, Wdt, bdt, Alog, Dp, Wout,
                                               Wg, bg, ws);

  float* gates = ws + 4 * 229120;   // 4 x (80 x 256)

  // Output tuple layout (floats): o4, o3, o2, o1, audio_flat
  const size_t o4_off = 0;                               // 80*256*49
  const size_t o3_off = 1003520;                         // 80*256*196
  const size_t o2_off = 5017600;                         // 80*256*784
  const size_t o1_off = 21073920;                        // 80*256*3136
  const size_t au_off = 85299200;                        // 20480

  // Stage 2: streaming gated residuals (float4, division-free indexing).
  gate_apply_vec4<<<dim3((784 + 255) / 256, 20480), 256, 0, stream>>>(
      (const float4*)x1, (float4*)(out + o1_off), gates + 0 * 20480, 784, 0);
  gate_apply_vec4<<<dim3(1, 20480), 256, 0, stream>>>(
      (const float4*)x2, (float4*)(out + o2_off), gates + 1 * 20480, 196, 1);
  gate_apply_vec4<<<dim3(1, 20480), 64, 0, stream>>>(
      (const float4*)x3, (float4*)(out + o3_off), gates + 2 * 20480, 49, 1);
  gate_apply_scalar<<<dim3(1, 20480), 64, 0, stream>>>(
      x4, out + o4_off, gates + 3 * 20480, 49, 1);   // HW=49 odd -> scalar

  copy_kernel<<<(20480 + 255) / 256, 256, 0, stream>>>(audio, out + au_off, 20480);
}
